// SparseMoE_29540785062131
// MI455X (gfx1250) — compile-verified, matrix operand order checked by reading, not run on hardware
//
#include <hip/hip_runtime.h>
#include <hip/hip_bf16.h>

// ---------------------------------------------------------------------------
// Sparse MoE (router + top-2 grouped SwiGLU experts) for gfx1250 / MI455X.
// bf16 WMMA (v_wmma_f32_16x16x32_bf16) with fp32 accumulation.
// GEMM2 A-tiles use GLOBAL_LOAD_ASYNC_TO_LDS_B128 (ASYNCcnt) since Hbuf is bf16.
// ---------------------------------------------------------------------------

typedef __attribute__((ext_vector_type(16))) __bf16 v16bf;
typedef __attribute__((ext_vector_type(8)))  float  v8f;

#define N_TOK 16384      // B*S
#define HDIM  1024
#define EXP   8
#define FDIM  4096
#define NPAIR (N_TOK * 2)

#define BM 128
#define BN 128
#define BK 32
#define SA 40            // padded LDS row stride (ushorts): 80B = 5*16B -> aligned, conflict-free
#define MAXT (NPAIR / BM + EXP)   // 264 worst-case M tiles across all experts

// header ints in ws: [0..7]=cnt  [8..16]=offsets  [17..25]=first_tile  [26..33]=cursor
#define H_CNT 0
#define H_OFF 8
#define H_FT  17
#define H_CUR 26

// fp32 -> bf16 via hardware convert (v_cvt_*bf16_f32), RNE
__device__ __forceinline__ unsigned short f2bf(float f) {
  union { __bf16 b; unsigned short u; } c;
  c.b = (__bf16)f;
  return c.u;
}

// Load one 16x32 bf16 A/B fragment row-slice from LDS per ISA 7.12.2 layout:
// lane<16 -> K{0..7,16..23}, lane>=16 -> K{8..15,24..31} of its row.
__device__ __forceinline__ v16bf ldfrag(const unsigned short* rowp, int half) {
  union { v16bf v; uint4 q[2]; } u;
  u.q[0] = *(const uint4*)(rowp + half * 8);
  u.q[1] = *(const uint4*)(rowp + 16 + half * 8);
  return u.v;
}

// ---------------------------------------------------------------------------
__global__ void moe_init_kernel(float* __restrict__ out, int* __restrict__ hdr) {
  size_t i = (size_t)blockIdx.x * blockDim.x + threadIdx.x;
  float4 z; z.x = z.y = z.z = z.w = 0.f;
  ((float4*)out)[i] = z;                      // grid sized to cover N_TOK*HDIM/4 exactly
  if (blockIdx.x == 0 && threadIdx.x < 64) hdr[threadIdx.x] = 0;
}

// One wave per token: logits = x @ Wr, raw logits out, top-2 + renormalized weights.
__global__ void moe_router_kernel(const float* __restrict__ X, const float* __restrict__ Wr,
                                  float* __restrict__ logits, int* __restrict__ ti,
                                  float* __restrict__ tw, int* __restrict__ hdr) {
  const int wave = threadIdx.x >> 5, lane = threadIdx.x & 31;
  const int t = blockIdx.x * 8 + wave;
  const float* xp = X + (size_t)t * HDIM;
  float acc[EXP];
#pragma unroll
  for (int j = 0; j < EXP; ++j) acc[j] = 0.f;
#pragma unroll 4
  for (int i = 0; i < HDIM / 32; ++i) {
    const int h = lane + (i << 5);
    const float xv = xp[h];
    const float4 wa = *(const float4*)(Wr + (size_t)h * EXP);
    const float4 wb = *(const float4*)(Wr + (size_t)h * EXP + 4);
    acc[0] += xv * wa.x; acc[1] += xv * wa.y; acc[2] += xv * wa.z; acc[3] += xv * wa.w;
    acc[4] += xv * wb.x; acc[5] += xv * wb.y; acc[6] += xv * wb.z; acc[7] += xv * wb.w;
  }
#pragma unroll
  for (int j = 0; j < EXP; ++j)
#pragma unroll
    for (int off = 16; off > 0; off >>= 1) acc[j] += __shfl_xor(acc[j], off);
  if (lane < EXP) logits[(size_t)t * EXP + lane] = acc[lane];
  if (lane == 0) {
    int i1 = 0;
#pragma unroll
    for (int j = 1; j < EXP; ++j) if (acc[j] > acc[i1]) i1 = j;
    int i2 = (i1 == 0) ? 1 : 0;
#pragma unroll
    for (int j = 0; j < EXP; ++j) if (j != i1 && acc[j] > acc[i2]) i2 = j;
    // renormalized top-2 softmax weights: full-softmax denominator cancels
    const float e2 = __expf(acc[i2] - acc[i1]);
    const float inv = 1.f / (1.f + e2);
    ti[t * 2] = i1;     tw[t * 2] = inv;
    ti[t * 2 + 1] = i2; tw[t * 2 + 1] = e2 * inv;
    atomicAdd(&hdr[H_CNT + i1], 1);
    atomicAdd(&hdr[H_CNT + i2], 1);
  }
}

__global__ void moe_scan_kernel(int* __restrict__ hdr) {
  if (threadIdx.x != 0) return;
  int off = 0, ft = 0;
  hdr[H_OFF] = 0; hdr[H_FT] = 0;
  for (int e = 0; e < EXP; ++e) {
    hdr[H_CUR + e] = off;
    off += hdr[H_CNT + e];
    ft  += (hdr[H_CNT + e] + BM - 1) / BM;
    hdr[H_OFF + 1 + e] = off;
    hdr[H_FT + 1 + e] = ft;
  }
}

__global__ void moe_scatter_kernel(const int* __restrict__ ti, const float* __restrict__ tw,
                                   int* __restrict__ hdr, int* __restrict__ tok_pair,
                                   float* __restrict__ w_pair) {
  const int t = blockIdx.x * blockDim.x + threadIdx.x;
#pragma unroll
  for (int k = 0; k < 2; ++k) {
    const int e = ti[t * 2 + k];
    const int pos = atomicAdd(&hdr[H_CUR + e], 1);
    tok_pair[pos] = t;
    w_pair[pos] = tw[t * 2 + k];
  }
}

// ---------------------------------------------------------------------------
// GEMM1: Hbuf[pair, :] = silu(x @ W1[e]) * (x @ W3[e])   (bf16 out)
// ---------------------------------------------------------------------------
__global__ __launch_bounds__(256) void moe_gemm1_kernel(
    const float* __restrict__ X, const float* __restrict__ W1, const float* __restrict__ W3,
    const int* __restrict__ hdr, const int* __restrict__ tok_pair,
    unsigned short* __restrict__ Hbuf) {
  __shared__ unsigned short As[2][BM][SA];
  __shared__ unsigned short B1[2][BN][SA];
  __shared__ unsigned short B3[2][BN][SA];

  const int tileId = blockIdx.x;
  const int* ft = hdr + H_FT;
  if (tileId >= ft[EXP]) return;                        // uniform exit: EXEC all-ones below
  int e = 0;
  while (!(tileId >= ft[e] && tileId < ft[e + 1])) ++e; // <=8 iters
  const int rowStart = hdr[H_OFF + e] + (tileId - ft[e]) * BM;
  const int rowEnd   = hdr[H_OFF + e + 1];
  const int nb = blockIdx.y * BN;
  const float* W1e = W1 + (size_t)e * HDIM * FDIM;
  const float* W3e = W3 + (size_t)e * HDIM * FDIM;

  const int tid = threadIdx.x;
  const int lane = tid & 31, wave = tid >> 5;
  const int m0 = (wave >> 2) * 64, n0 = (wave & 3) * 32;
  const int half = lane >> 4, l16 = lane & 15;

  const int kgA = tid & 7,  r0A = tid >> 3;   // A tile: [r0A+32p][kgA*4..+3]
  const int ngB = tid & 31, k0B = tid >> 5;   // B tile: [k0B+8p][ngB*4..+3]

  int tok[4];
#pragma unroll
  for (int p = 0; p < 4; ++p) {
    const int pr = rowStart + r0A + 32 * p;
    tok[p] = tok_pair[pr < rowEnd ? pr : rowStart];
  }

  float4 ra[4], rb1[4], rb3[4];
  auto gload = [&](int kt) {
    const int k0 = kt * BK;
#pragma unroll
    for (int p = 0; p < 4; ++p)
      ra[p] = *(const float4*)(X + (size_t)tok[p] * HDIM + k0 + kgA * 4);
#pragma unroll
    for (int p = 0; p < 4; ++p) {
      const size_t ro = (size_t)(k0 + k0B + 8 * p) * FDIM + nb + ngB * 4;
      rb1[p] = *(const float4*)(W1e + ro);
      rb3[p] = *(const float4*)(W3e + ro);
    }
    // keep the L2->WGP pipe primed for the weight streams (global_prefetch_b8)
    if (kt + 2 < HDIM / BK) {
      const size_t pf = (size_t)((kt + 2) * BK + k0B) * FDIM + nb + ngB * 4;
      __builtin_prefetch(W1e + pf, 0, 3);
      __builtin_prefetch(W3e + pf, 0, 3);
    }
  };
  auto sstore = [&](int buf) {
#pragma unroll
    for (int p = 0; p < 4; ++p) {
      ushort4 s;
      s.x = f2bf(ra[p].x); s.y = f2bf(ra[p].y); s.z = f2bf(ra[p].z); s.w = f2bf(ra[p].w);
      *(ushort4*)&As[buf][r0A + 32 * p][kgA * 4] = s;
    }
#pragma unroll
    for (int p = 0; p < 4; ++p) {
      const int k = k0B + 8 * p;
      B1[buf][ngB * 4 + 0][k] = f2bf(rb1[p].x);
      B1[buf][ngB * 4 + 1][k] = f2bf(rb1[p].y);
      B1[buf][ngB * 4 + 2][k] = f2bf(rb1[p].z);
      B1[buf][ngB * 4 + 3][k] = f2bf(rb1[p].w);
      B3[buf][ngB * 4 + 0][k] = f2bf(rb3[p].x);
      B3[buf][ngB * 4 + 1][k] = f2bf(rb3[p].y);
      B3[buf][ngB * 4 + 2][k] = f2bf(rb3[p].z);
      B3[buf][ngB * 4 + 3][k] = f2bf(rb3[p].w);
    }
  };

  v8f accG[4][2] = {};
  v8f accU[4][2] = {};

  gload(0); sstore(0); __syncthreads();
  const int KT = HDIM / BK;
  for (int kt = 0; kt < KT; ++kt) {
    const int buf = kt & 1;
    if (kt + 1 < KT) gload(kt + 1);
    v16bf a[4], bg[2], bu[2];
#pragma unroll
    for (int mt = 0; mt < 4; ++mt) a[mt] = ldfrag(&As[buf][m0 + mt * 16 + l16][0], half);
#pragma unroll
    for (int nt = 0; nt < 2; ++nt) {
      bg[nt] = ldfrag(&B1[buf][n0 + nt * 16 + l16][0], half);
      bu[nt] = ldfrag(&B3[buf][n0 + nt * 16 + l16][0], half);
    }
#pragma unroll
    for (int mt = 0; mt < 4; ++mt)
#pragma unroll
      for (int nt = 0; nt < 2; ++nt) {
        accG[mt][nt] = __builtin_amdgcn_wmma_f32_16x16x32_bf16(
            false, a[mt], false, bg[nt], (short)0, accG[mt][nt], false, false);
        accU[mt][nt] = __builtin_amdgcn_wmma_f32_16x16x32_bf16(
            false, a[mt], false, bu[nt], (short)0, accU[mt][nt], false, false);
      }
    __syncthreads();
    if (kt + 1 < KT) { sstore((kt + 1) & 1); __syncthreads(); }
  }

  // SwiGLU epilogue -> bf16 Hbuf rows (only rows inside this expert segment)
#pragma unroll
  for (int mt = 0; mt < 4; ++mt) {
#pragma unroll
    for (int i = 0; i < 8; ++i) {
      const int pr = rowStart + m0 + mt * 16 + i + half * 8;
      if (pr < rowEnd) {
        unsigned short* dst = Hbuf + (size_t)pr * FDIM + nb + n0 + l16;
#pragma unroll
        for (int nt = 0; nt < 2; ++nt) {
          const float g = accG[mt][nt][i], u = accU[mt][nt][i];
          const float h = g * (1.f / (1.f + __expf(-g))) * u;
          dst[nt * 16] = f2bf(h);
        }
      }
    }
  }
}

// ---------------------------------------------------------------------------
// GEMM2: out[tok] += (Hbuf[pair] @ W2[e]) * w_pair   (fp32 atomics scatter)
// A-tiles (bf16 Hbuf) stream global->LDS via GLOBAL_LOAD_ASYNC_TO_LDS_B128.
// ---------------------------------------------------------------------------
__global__ __launch_bounds__(256) void moe_gemm2_kernel(
    const unsigned short* __restrict__ Hbuf, const float* __restrict__ W2,
    const int* __restrict__ hdr, const int* __restrict__ tok_pair,
    const float* __restrict__ w_pair, float* __restrict__ out) {
  __shared__ unsigned short As[2][BM][SA];
  __shared__ unsigned short Bs[2][BN][SA];

  const int tileId = blockIdx.x;
  const int* ft = hdr + H_FT;
  if (tileId >= ft[EXP]) return;
  int e = 0;
  while (!(tileId >= ft[e] && tileId < ft[e + 1])) ++e;
  const int rowStart = hdr[H_OFF + e] + (tileId - ft[e]) * BM;
  const int rowEnd   = hdr[H_OFF + e + 1];
  const int nb = blockIdx.y * BN;
  const float* W2e = W2 + (size_t)e * FDIM * HDIM;

  const int tid = threadIdx.x;
  const int lane = tid & 31, wave = tid >> 5;
  const int m0 = (wave >> 2) * 64, n0 = (wave & 3) * 32;
  const int half = lane >> 4, l16 = lane & 15;

  const int ngB = tid & 31, k0B = tid >> 5;

  const unsigned ldsA = (unsigned)(size_t)&As[0][0][0];

  // Async copy of the next A tile straight into LDS (no VGPR staging):
  // 128 rows x 64B; 4 lanes of 16B per row; 2 issues of 256 lanes.
  auto issueA = [&](int kt, int buf) {
    const int k0 = kt * BK;
    const int q = tid & 3, rr = tid >> 2;
#pragma unroll
    for (int p = 0; p < 2; ++p) {
      const int r = rr + 64 * p;
      const unsigned short* g = Hbuf + (size_t)(rowStart + r) * FDIM + k0 + q * 8;
      const unsigned lds = ldsA + (unsigned)(((buf * BM + r) * SA + q * 8) * 2);
      asm volatile("global_load_async_to_lds_b128 %0, %1, off"
                   :: "v"(lds), "v"(g) : "memory");
    }
  };

  float4 rb[4];
  auto gloadB = [&](int kt) {
    const int k0 = kt * BK;
#pragma unroll
    for (int p = 0; p < 4; ++p)
      rb[p] = *(const float4*)(W2e + (size_t)(k0 + k0B + 8 * p) * HDIM + nb + ngB * 4);
    if (kt + 2 < FDIM / BK) {
      const size_t pf = (size_t)((kt + 2) * BK + k0B) * HDIM + nb + ngB * 4;
      __builtin_prefetch(W2e + pf, 0, 3);
    }
  };
  auto sstoreB = [&](int buf) {
#pragma unroll
    for (int p = 0; p < 4; ++p) {
      const int k = k0B + 8 * p;
      Bs[buf][ngB * 4 + 0][k] = f2bf(rb[p].x);
      Bs[buf][ngB * 4 + 1][k] = f2bf(rb[p].y);
      Bs[buf][ngB * 4 + 2][k] = f2bf(rb[p].z);
      Bs[buf][ngB * 4 + 3][k] = f2bf(rb[p].w);
    }
  };

  v8f acc[4][2] = {};

  issueA(0, 0);
  gloadB(0); sstoreB(0);
  asm volatile("s_wait_asynccnt 0" ::: "memory");
  __syncthreads();

  const int KT = FDIM / BK;
  for (int kt = 0; kt < KT; ++kt) {
    const int buf = kt & 1;
    // Safe to issue into buf^1 here: previous iteration's trailing barrier
    // guarantees every wave finished reading buf^1.
    if (kt + 1 < KT) { issueA(kt + 1, buf ^ 1); gloadB(kt + 1); }
    v16bf a[4], b[2];
#pragma unroll
    for (int mt = 0; mt < 4; ++mt) a[mt] = ldfrag(&As[buf][m0 + mt * 16 + l16][0], half);
#pragma unroll
    for (int nt = 0; nt < 2; ++nt) b[nt] = ldfrag(&Bs[buf][n0 + nt * 16 + l16][0], half);
#pragma unroll
    for (int mt = 0; mt < 4; ++mt)
#pragma unroll
      for (int nt = 0; nt < 2; ++nt)
        acc[mt][nt] = __builtin_amdgcn_wmma_f32_16x16x32_bf16(
            false, a[mt], false, b[nt], (short)0, acc[mt][nt], false, false);
    __syncthreads();
    if (kt + 1 < KT) {
      sstoreB(buf ^ 1);
      asm volatile("s_wait_asynccnt 0" ::: "memory");   // A tile landed in LDS
      __syncthreads();
    }
  }

  // weighted scatter into out
#pragma unroll
  for (int mt = 0; mt < 4; ++mt) {
#pragma unroll
    for (int i = 0; i < 8; ++i) {
      const int pr = rowStart + m0 + mt * 16 + i + half * 8;
      if (pr < rowEnd) {
        const int tk = tok_pair[pr];
        const float wv = w_pair[pr];
        float* dst = out + (size_t)tk * HDIM + nb + n0 + l16;
#pragma unroll
        for (int nt = 0; nt < 2; ++nt)
          atomicAdd(dst + nt * 16, acc[mt][nt][i] * wv);
      }
    }
  }
}

// ---------------------------------------------------------------------------
extern "C" void kernel_launch(void* const* d_in, const int* in_sizes, int n_in,
                              void* d_out, int out_size, void* d_ws, size_t ws_size,
                              hipStream_t stream) {
  const float* X  = (const float*)d_in[0];
  const float* Wr = (const float*)d_in[1];
  const float* W1 = (const float*)d_in[2];
  const float* W2 = (const float*)d_in[3];
  const float* W3 = (const float*)d_in[4];

  float* out    = (float*)d_out;                       // [N_TOK*HDIM]
  float* logits = out + (size_t)N_TOK * HDIM;          // [N_TOK*EXP]

  // workspace layout
  char* w = (char*)d_ws;
  size_t off = 1024;                                    // header (64 ints) + pad
  int*   hdr      = (int*)w;
  int*   ti       = (int*)(w + off);   off += (size_t)N_TOK * 2 * sizeof(int);
  float* tw       = (float*)(w + off); off += (size_t)N_TOK * 2 * sizeof(float);
  int*   tok_pair = (int*)(w + off);   off += (size_t)(NPAIR + 128) * sizeof(int);
  float* w_pair   = (float*)(w + off); off += (size_t)(NPAIR + 128) * sizeof(float);
  off = (off + 255) & ~(size_t)255;
  unsigned short* Hbuf = (unsigned short*)(w + off);    // [(NPAIR+128) x FDIM] bf16

  moe_init_kernel<<<(N_TOK * (HDIM / 4)) / 256, 256, 0, stream>>>(out, hdr);
  moe_router_kernel<<<N_TOK / 8, 256, 0, stream>>>(X, Wr, logits, ti, tw, hdr);
  moe_scan_kernel<<<1, 32, 0, stream>>>(hdr);
  moe_scatter_kernel<<<N_TOK / 256, 256, 0, stream>>>(ti, tw, hdr, tok_pair, w_pair);
  moe_gemm1_kernel<<<dim3(MAXT, FDIM / BN), 256, 0, stream>>>(X, W1, W3, hdr, tok_pair, Hbuf);
  moe_gemm2_kernel<<<dim3(MAXT, HDIM / BN), 256, 0, stream>>>(Hbuf, W2, hdr, tok_pair, w_pair, out);
}
